// AIMv2FlashAttention2_84825604096686
// MI455X (gfx1250) — matrix operand order repaired and checked by reading, not compiled, not run on hardware
//
#include <hip/hip_runtime.h>
#include <hip/hip_bf16.h>
#include <stdint.h>

typedef uint32_t u32;
typedef uint16_t u16;
typedef __attribute__((ext_vector_type(4)))  u32    u32x4;
typedef __attribute__((ext_vector_type(4)))  int    i32x4;
typedef __attribute__((ext_vector_type(8)))  float  v8f;
typedef __attribute__((ext_vector_type(16))) __bf16 bf16x16;

union Frag { u32x4 q[2]; u32 u[8]; bf16x16 b; };
union Acc  { v8f v; float f[8]; };

__device__ __forceinline__ u16 f2bf(float f) {
  u32 u = __float_as_uint(f);
  u32 r = u + 0x7FFFu + ((u >> 16) & 1u);   // round-to-nearest-even
  return (u16)(r >> 16);
}
__device__ __forceinline__ float bf2f(u16 h) {
  return __uint_as_float(((u32)h) << 16);
}

__device__ __forceinline__ v8f wmma_bf16(const Frag& a, const Frag& b, v8f c) {
  return __builtin_amdgcn_wmma_f32_16x16x32_bf16(false, a.b, false, b.b,
                                                 (short)0, c, false, false);
}

// ---- DPP16 row (16-lane) reductions: xor1, xor2, xor7, xor15 ---------------
__device__ __forceinline__ float fmaxf_(float a, float b) { return fmaxf(a, b); }
__device__ __forceinline__ float faddf_(float a, float b) { return a + b; }

#define DPP_STEP(op, x, ctrl)                                                  \
  do {                                                                         \
    float _t = __int_as_float(__builtin_amdgcn_update_dpp(                     \
        0, __float_as_int(x), ctrl, 0xF, 0xF, true));                          \
    x = op(x, _t);                                                             \
  } while (0)

__device__ __forceinline__ float row16_max(float x) {
  DPP_STEP(fmaxf_, x, 0xB1);   // quad_perm [1,0,3,2]  (xor 1)
  DPP_STEP(fmaxf_, x, 0x4E);   // quad_perm [2,3,0,1]  (xor 2)
  DPP_STEP(fmaxf_, x, 0x141);  // row_half_mirror      (xor 7 in 8)
  DPP_STEP(fmaxf_, x, 0x140);  // row_mirror           (xor 15 in 16)
  return x;
}
__device__ __forceinline__ float row16_sum(float x) {
  DPP_STEP(faddf_, x, 0xB1);
  DPP_STEP(faddf_, x, 0x4E);
  DPP_STEP(faddf_, x, 0x141);
  DPP_STEP(faddf_, x, 0x140);
  return x;
}

// ---- async 16B global -> LDS copy (ASYNCcnt path), guarded -----------------
__device__ __forceinline__ void async_ld16(const u16* g, u16* l) {
#if __has_builtin(__builtin_amdgcn_global_load_async_to_lds_b128)
  __builtin_amdgcn_global_load_async_to_lds_b128(
      (__attribute__((address_space(1))) i32x4*)(u16*)g,
      (__attribute__((address_space(3))) i32x4*)l, 0, 0);
#else
  *(u32x4*)l = *(const u32x4*)g;   // fallback: reg-mediated copy
#endif
}
__device__ __forceinline__ void wait_async0() {
  asm volatile("s_wait_asynccnt 0x0" ::: "memory");
}

// ---------------------------------------------------------------------------
__global__ __launch_bounds__(256) void cvt_kernel(const float* __restrict__ in,
                                                  u16* __restrict__ out, long n) {
  long i = (long)blockIdx.x * 256 + threadIdx.x;
  if (i < n) out[i] = f2bf(in[i]);
}

// fp32 [K,N] row-major -> bf16 transposed [N,K] row-major
__global__ __launch_bounds__(256) void cvtT_kernel(const float* __restrict__ in,
                                                   u16* __restrict__ out,
                                                   int K, int N) {
  long i = (long)blockIdx.x * 256 + threadIdx.x;
  if (i >= (long)K * N) return;
  int k = (int)(i % K);
  int n = (int)(i / K);
  out[i] = f2bf(in[(long)k * N + n]);
}

// ---------------------------------------------------------------------------
// GEMM: C[M,N] = A[M,K](bf16) * Bt[N,K]^T(bf16) + bias ; per-wave tile 64x64
template <bool OUT_BF16>
__global__ __launch_bounds__(256)
void gemm64x64(const u16* __restrict__ A, const u16* __restrict__ Bt,
               const float* __restrict__ bias, void* __restrict__ Cout,
               int N, int K) {
  const int lane = threadIdx.x & 31;
  const int ln = lane & 15, lh = lane >> 4;
  const int wave = blockIdx.x * 8 + (threadIdx.x >> 5);
  const int wavesN = N >> 6;
  const int m0 = (wave / wavesN) << 6;
  const int n0 = (wave % wavesN) << 6;

  Acc acc[4][4];
#pragma unroll
  for (int mt = 0; mt < 4; ++mt)
#pragma unroll
    for (int nt = 0; nt < 4; ++nt)
#pragma unroll
      for (int i = 0; i < 8; ++i) acc[mt][nt].f[i] = 0.f;

  const u16* Abase = A + (size_t)(m0 + ln) * K + lh * 8;
  const u16* Bbase = Bt + (size_t)(n0 + ln) * K + lh * 8;

  for (int k0 = 0; k0 < K; k0 += 32) {
    Frag a[4], bb[4];
#pragma unroll
    for (int mt = 0; mt < 4; ++mt) {
      const u16* p = Abase + (size_t)(mt * 16) * K + k0;
      a[mt].q[0] = *(const u32x4*)(p);        // K = lh*8 + 0..7
      a[mt].q[1] = *(const u32x4*)(p + 16);   // K = 16 + lh*8 + 0..7
    }
#pragma unroll
    for (int nt = 0; nt < 4; ++nt) {
      const u16* p = Bbase + (size_t)(nt * 16) * K + k0;
      bb[nt].q[0] = *(const u32x4*)(p);       // K = lh*8 + 0..7
      bb[nt].q[1] = *(const u32x4*)(p + 16);  // K = 16 + lh*8 + 0..7
    }
    if (k0 + 32 < K) {
      __builtin_prefetch(Abase + k0 + 32, 0, 3);
      __builtin_prefetch(Bbase + k0 + 32, 0, 3);
    }
#pragma unroll
    for (int mt = 0; mt < 4; ++mt)
#pragma unroll
      for (int nt = 0; nt < 4; ++nt)
        acc[mt][nt].v = wmma_bf16(a[mt], bb[nt], acc[mt][nt].v);
  }

#pragma unroll
  for (int nt = 0; nt < 4; ++nt) {
    int c = n0 + nt * 16 + ln;
    float bv = bias[c];
#pragma unroll
    for (int mt = 0; mt < 4; ++mt)
#pragma unroll
      for (int r = 0; r < 8; ++r) {
        int mrow = m0 + mt * 16 + r + 8 * lh;
        float val = acc[mt][nt].f[r] + bv;
        if (OUT_BF16) ((u16*)Cout)[(size_t)mrow * N + c] = f2bf(val);
        else          ((float*)Cout)[(size_t)mrow * N + c] = val;
      }
  }
}

// ---------------------------------------------------------------------------
// RoPE + softmax prescale (on q) + reshuffle [T,3072] -> q/k/v [B*H, L, D]
__global__ __launch_bounds__(256)
void rope_kernel(const u16* __restrict__ qkv, const float* __restrict__ cosb,
                 const float* __restrict__ sinb, u16* __restrict__ qh,
                 u16* __restrict__ kh, u16* __restrict__ vh) {
  long idx = (long)blockIdx.x * 256 + threadIdx.x;
  if (idx >= 3L * 8192 * 1024) return;
  int s  = (int)(idx / 8388608L);
  long rem = idx % 8388608L;
  int t = (int)(rem >> 10), f = (int)(rem & 1023);
  int h = f >> 6, d = f & 63;
  int b = t >> 10, li = t & 1023;

  float x = bf2f(qkv[(size_t)t * 3072 + s * 1024 + f]);
  float val;
  if (s == 2) {
    val = x;
  } else {
    int d2 = d & 31;
    bool up = d >= 32;
    int pf = s * 1024 + h * 64 + (up ? d - 32 : d + 32);
    float xp = bf2f(qkv[(size_t)t * 3072 + pf]);
    float c = cosb[(size_t)t * 64 + d2];
    float sn = sinb[(size_t)t * 64 + d2];
    val = up ? (x * c + xp * sn) : (x * c - xp * sn);
    if (s == 0) val *= 0.125f;                 // fold 1/sqrt(D) into q
  }
  size_t dst = (((size_t)b * 16 + h) * 1024 + li) * 64 + d;
  u16* outp = (s == 0) ? qh : (s == 1) ? kh : vh;
  outp[dst] = f2bf(val);
}

// ---------------------------------------------------------------------------
// Flash attention. Block = 8 waves sharing one (b,h); each wave owns a 16-row
// q tile. Per 32-step, the block cooperatively async-copies the 32x64 K and V
// tiles into LDS (ASYNCcnt), then everyone computes from LDS.
#define KV_PITCH 72   // u16 elements per LDS tile row (144B: 16B aligned, bank-skewed)

__global__ __launch_bounds__(256)
void flash_kernel(const u16* __restrict__ qh, const u16* __restrict__ kh,
                  const u16* __restrict__ vh, u16* __restrict__ attn) {
  __shared__ u16 kt[32 * KV_PITCH];
  __shared__ u16 vt[32 * KV_PITCH];
  __shared__ u16 plds[8][16 * 34];

  const int tid = threadIdx.x;
  const int lane = tid & 31, ln = lane & 15, lh = lane >> 4, wv = tid >> 5;
  const int bh = blockIdx.x >> 3;
  const int i0 = ((blockIdx.x & 7) << 7) + (wv << 4);
  const int b = bh >> 4, h = bh & 15;

  const u16* q  = qh + (size_t)bh * 65536;
  const u16* kp = kh + (size_t)bh * 65536;
  const u16* vp = vh + (size_t)bh * 65536;

  // cooperative staging role: 256 threads x (16B K + 16B V) = 32x64 bf16 tiles
  const int srow = tid >> 3, schunk = tid & 7;
  const u16* ksrc = kp + (size_t)srow * 64 + schunk * 8;
  const u16* vsrc = vp + (size_t)srow * 64 + schunk * 8;
  u16* kdst = &kt[srow * KV_PITCH + schunk * 8];
  u16* vdst = &vt[srow * KV_PITCH + schunk * 8];

  Frag qa[2];                                // Q tile 16x64: two 16x32 A frags
  {
    const u16* p = q + (size_t)(i0 + ln) * 64 + lh * 8;
    qa[0].q[0] = *(const u32x4*)(p);
    qa[0].q[1] = *(const u32x4*)(p + 16);
    qa[1].q[0] = *(const u32x4*)(p + 32);
    qa[1].q[1] = *(const u32x4*)(p + 48);
  }

  Acc o[4];
  float m[8], l[8];
#pragma unroll
  for (int nt = 0; nt < 4; ++nt)
#pragma unroll
    for (int r = 0; r < 8; ++r) o[nt].f[r] = 0.f;
#pragma unroll
  for (int r = 0; r < 8; ++r) { m[r] = -1e30f; l[r] = 0.f; }

  for (int j0 = 0; j0 < 1024; j0 += 32) {
    // stage K/V tiles for this step (ASYNCcnt-tracked DMA into LDS)
    async_ld16(ksrc + (size_t)j0 * 64, kdst);
    async_ld16(vsrc + (size_t)j0 * 64, vdst);
    wait_async0();
    __syncthreads();                          // tiles visible to all waves

    // S = Q * K^T  (two 16x16 C tiles, K-dim = 64) -- B frags from LDS
    Acc s[2];
#pragma unroll
    for (int jt = 0; jt < 2; ++jt) {
#pragma unroll
      for (int i = 0; i < 8; ++i) s[jt].f[i] = 0.f;
      Frag kb;
      const u16* p = &kt[(jt * 16 + ln) * KV_PITCH + lh * 8];
      kb.q[0] = *(const u32x4*)(p);
      kb.q[1] = *(const u32x4*)(p + 16);
      s[jt].v = wmma_bf16(qa[0], kb, s[jt].v);
      kb.q[0] = *(const u32x4*)(p + 32);
      kb.q[1] = *(const u32x4*)(p + 48);
      s[jt].v = wmma_bf16(qa[1], kb, s[jt].v);
    }

    // online softmax: DPP16 row reductions (no LDS traffic)
    float alpha[8];
#pragma unroll
    for (int r = 0; r < 8; ++r) {
      float mx = row16_max(fmaxf(s[0].f[r], s[1].f[r]));
      float mn = fmaxf(m[r], mx);
      alpha[r] = __expf(m[r] - mn);
      m[r] = mn;
      s[0].f[r] = __expf(s[0].f[r] - mn);
      s[1].f[r] = __expf(s[1].f[r] - mn);
      float rs = row16_sum(s[0].f[r] + s[1].f[r]);
      l[r] = l[r] * alpha[r] + rs;
#pragma unroll
      for (int nt = 0; nt < 4; ++nt) o[nt].f[r] *= alpha[r];
    }

    // P: C layout -> A layout via per-wave LDS staging
#pragma unroll
    for (int r = 0; r < 8; ++r) {
      int row = r + 8 * lh;
      plds[wv][row * 34 + ln]      = f2bf(s[0].f[r]);
      plds[wv][row * 34 + 16 + ln] = f2bf(s[1].f[r]);
    }
    asm volatile("s_wait_dscnt 0x0" ::: "memory");
    Frag pa;
#pragma unroll
    for (int v = 0; v < 4; ++v) {
      int kkk = lh * 8 + 2 * v;
      pa.u[v]     = (u32)plds[wv][ln * 34 + kkk]
                  | ((u32)plds[wv][ln * 34 + kkk + 1] << 16);
      pa.u[v + 4] = (u32)plds[wv][ln * 34 + 16 + kkk]
                  | ((u32)plds[wv][ln * 34 + 16 + kkk + 1] << 16);
    }

    // O += P(16x32) * V(32x64): V B-frags gathered from the LDS tile
#pragma unroll
    for (int nt = 0; nt < 4; ++nt) {
      Frag vb;
      int d = nt * 16 + ln;
#pragma unroll
      for (int v = 0; v < 8; ++v) {
        int jr = lh * 8 + (v >> 2) * 16 + 2 * (v & 3);
        u32 lo = vt[jr * KV_PITCH + d];
        u32 hi = vt[(jr + 1) * KV_PITCH + d];
        vb.u[v] = lo | (hi << 16);
      }
      o[nt].v = wmma_bf16(pa, vb, o[nt].v);
    }
    __syncthreads();                          // all waves done with tiles
  }

#pragma unroll
  for (int r = 0; r < 8; ++r) {
    float inv = 1.0f / l[r];
    int t = (b << 10) + i0 + r + 8 * lh;
#pragma unroll
    for (int nt = 0; nt < 4; ++nt)
      attn[(size_t)t * 1024 + h * 64 + nt * 16 + ln] = f2bf(o[nt].f[r] * inv);
  }
}

// ---------------------------------------------------------------------------
extern "C" void kernel_launch(void* const* d_in, const int* in_sizes, int n_in,
                              void* d_out, int out_size, void* d_ws, size_t ws_size,
                              hipStream_t stream) {
  (void)in_sizes; (void)n_in; (void)out_size; (void)ws_size;
  const float* hidden = (const float*)d_in[0];
  const float* cosb   = (const float*)d_in[1];
  const float* sinb   = (const float*)d_in[2];
  const float* w_qkv  = (const float*)d_in[3];
  const float* b_qkv  = (const float*)d_in[4];
  const float* w_proj = (const float*)d_in[5];
  const float* b_proj = (const float*)d_in[6];
  // d_in[7] (cu_seqlens) unused: uniform L=1024

  char* ws = (char*)d_ws;
  u16* hiddenB = (u16*)(ws);                        // 16 MB; reused as qh later
  u16* wqkvT   = (u16*)(ws + 16777216);             //  6 MB
  u16* wprojT  = (u16*)(ws + 23068672);             //  2 MB
  u16* qkvB    = (u16*)(ws + 25165824);             // 50 MB; reused as attn later
  u16* khp     = (u16*)(ws + 75497472);             // 16 MB
  u16* vhp     = (u16*)(ws + 92274688);             // 16 MB
  u16* qhp  = hiddenB;                              // alias (hiddenB dead after GEMM1)
  u16* attn = qkvB;                                 // alias (qkvB dead after RoPE)

  cvt_kernel <<<32768, 256, 0, stream>>>(hidden, hiddenB, 8388608L);
  cvtT_kernel<<<12288, 256, 0, stream>>>(w_qkv, wqkvT, 1024, 3072);
  cvtT_kernel<<<4096,  256, 0, stream>>>(w_proj, wprojT, 1024, 1024);

  // QKV projection: (8192/64)*(3072/64) = 6144 waves / 8 per block
  gemm64x64<true><<<768, 256, 0, stream>>>(hiddenB, wqkvT, b_qkv, qkvB, 3072, 1024);

  rope_kernel<<<98304, 256, 0, stream>>>(qkvB, cosb, sinb, qhp, khp, vhp);

  // 128 heads * 8 q-chunks = 1024 blocks (8 waves each, one q-tile per wave)
  flash_kernel<<<1024, 256, 0, stream>>>(qhp, khp, vhp, attn);

  // Output projection: (8192/64)*(1024/64) = 2048 waves / 8 per block
  gemm64x64<false><<<256, 256, 0, stream>>>(attn, wprojT, b_proj, d_out, 1024, 1024);
}
